// JacobiKANLayer_70866960384498
// MI455X (gfx1250) — matrix operand order repaired and checked by reading, not compiled, not run on hardware
//
#include <hip/hip_runtime.h>
#include <hip/hip_bf16.h>

// ---------------------------------------------------------------------------
// JacobiKAN layer for MI455X (gfx1250, wave32, WMMA).
//   z = silu(x) @ W^T + sum_{d=0..8} P_d(tanh(x)) @ J[:,:,d] -> LayerNorm -> SiLU
// P0 == 1 folded into a per-column bias => fused GEMM with K = 9*1024
// (branch 0 = silu path, branches 1..8 = Jacobi P1..P8) via
// v_wmma_f32_16x16x32_bf16 with f32 accumulation.
// Block tile 128x128 to halve L2 B-traffic; activations staged in 90 KB
// dynamic LDS (gfx1250 allows 320 KB/WG).
// ---------------------------------------------------------------------------

#define IN_F   1024
#define OUT_F  1024
#define BATCH  8192
#define NBR    9                    // silu + P1..P8  (P0 folded into bias)
#define KTILES (NBR * IN_F / 32)    // 288 k-tiles of 32
#define BM     128
#define ACT_STRIDE 40               // halves per row: 80B, 16B-aligned, conflict-free
#define ACT_BYTES  (NBR * BM * ACT_STRIDE * 2)   // 92160

typedef __attribute__((ext_vector_type(16))) __bf16 v16bf;
typedef __attribute__((ext_vector_type(8)))  float  v8f;

union FragAB { uint4 u[2]; v16bf v; };

__device__ __forceinline__ unsigned short f2bf(float f) {
    unsigned int u = __float_as_uint(f);
    unsigned int r = u + 0x7FFFu + ((u >> 16) & 1u);   // round-to-nearest-even
    return (unsigned short)(r >> 16);
}
__device__ __forceinline__ float fast_rcp(float f) { return __builtin_amdgcn_rcpf(f); }
__device__ __forceinline__ float fast_tanh(float x) {
    // tanh(x) = 1 - 2/(e^{2x}+1); exact limits at +/-inf, TRANS ops only
    return 1.0f - 2.0f * fast_rcp(__expf(2.0f * x) + 1.0f);
}
__device__ __forceinline__ float fast_silu(float x) {
    return x * fast_rcp(1.0f + __expf(-x));
}

// ---------------------------------------------------------------------------
// Kernel 1: pack base_weights + jacobi_coeff (orders 1..8) into fragment-
// native bf16 layout: wb[ ((kt*64 + ntile)*2 + h)*32 + lane ] : uint4.
// B operand (32x16, KxN), contiguous k order used here: lane n=l%16; lanes<16 hold K=0..15, lanes>=16 hold
// K=16..31; h selects first/second 8 K's of the lane.
// ---------------------------------------------------------------------------
__global__ __launch_bounds__(256) void kan_pack_b(
    const float* __restrict__ bw,   // [OUT_F][IN_F]
    const float* __restrict__ jc,   // [IN_F][OUT_F][9]
    uint4* __restrict__ wb)
{
    unsigned int idx = blockIdx.x * 256u + threadIdx.x;   // < KTILES*4096
    int lane = idx & 31;
    int h    = (idx >> 5) & 1;
    int nt   = (idx >> 6) & 63;
    int kt   = idx >> 12;
    if (kt >= KTILES) return;

    int n     = nt * 16 + (lane & 15);
    int kbase = ((lane < 16) ? 0 : 16) + h * 8;
    int d     = kt >> 5;            // 0 = silu/base weights, 1..8 = P1..P8
    int ibase = (kt & 31) * 32;

    unsigned short hh[8];
#pragma unroll
    for (int e = 0; e < 8; ++e) {
        int i = ibase + kbase + e;
        float f = (d == 0) ? bw[(size_t)n * IN_F + i]
                           : jc[(size_t)i * (OUT_F * 9) + (size_t)n * 9 + d];
        hh[e] = f2bf(f);
    }
    uint4 o;
    o.x = (unsigned)hh[0] | ((unsigned)hh[1] << 16);
    o.y = (unsigned)hh[2] | ((unsigned)hh[3] << 16);
    o.z = (unsigned)hh[4] | ((unsigned)hh[5] << 16);
    o.w = (unsigned)hh[6] | ((unsigned)hh[7] << 16);
    wb[idx] = o;
}

// ---------------------------------------------------------------------------
// Kernel 2: per-column bias from the P0 branch: jbias[o] = sum_i jc[i][o][0]
// ---------------------------------------------------------------------------
__global__ __launch_bounds__(256) void kan_jbias(
    const float* __restrict__ jc, float* __restrict__ jb)
{
    int o = blockIdx.x * 256 + threadIdx.x;   // 1024 threads total
    float s = 0.0f;
    for (int i = 0; i < IN_F; ++i)
        s += jc[(size_t)i * (OUT_F * 9) + (size_t)o * 9];
    jb[o] = s;
}

// ---------------------------------------------------------------------------
// Kernel 3: fused activation-generation + 9-branch GEMM.
// Block tile 128(M) x 128(N), 256 threads = 8 waves as 4(M) x 2(N);
// each wave: 2x4 16x16 f32 accumulators.
// ---------------------------------------------------------------------------
__global__ __launch_bounds__(256) void kan_gemm(
    const float* __restrict__ x,    // [BATCH][IN_F]
    const uint4* __restrict__ wb,   // packed B fragments
    float* __restrict__ z)          // [BATCH][OUT_F]
{
    extern __shared__ __align__(16) unsigned short act[];   // [NBR][BM][ACT_STRIDE]

    const int tid  = threadIdx.x;
    const int lane = tid & 31;
    const int wave = tid >> 5;
    const int wm   = wave & 3;      // 0..3 -> 32 rows each
    const int wn   = wave >> 2;     // 0..1 -> 64 cols each
    const int rowBase = blockIdx.x * BM;
    const int ntBase  = blockIdx.y * 8;

    v8f acc[2][4];
#pragma unroll
    for (int mi = 0; mi < 2; ++mi)
#pragma unroll
        for (int nj = 0; nj < 4; ++nj)
#pragma unroll
            for (int e = 0; e < 8; ++e) acc[mi][nj][e] = 0.0f;

    const int ar = wave * 16;       // this thread fills rows ar..ar+15
    const int ac = lane;            // column within the 32-wide k-chunk
    const float* xp = x + (size_t)(rowBase + ar) * IN_F + ac;

    const int arow = wm * 32 + (lane & 15);
    const int kcol = (lane < 16) ? 0 : 16;           // contiguous k order (16*(lane/16) + s), same as B
    const size_t bWave = (size_t)(ntBase + wn * 4) * 64 + lane;

    for (int ic = 0; ic < IN_F / 32; ++ic) {
        // ---- generate activations for x-tile [128][32] ----
#pragma unroll
        for (int j = 0; j < 16; ++j) {
            float xv = xp[(size_t)j * IN_F + ic * 32];
            float t  = fast_tanh(xv);
            int r = ar + j;
            act[(0 * BM + r) * ACT_STRIDE + ac] = f2bf(fast_silu(xv));
            float pm2 = 1.0f;                        // P0
            float pm1 = 2.0f * t;                    // P1 (a=b=1)
            act[(1 * BM + r) * ACT_STRIDE + ac] = f2bf(pm1);
#pragma unroll
            for (int i = 2; i <= 8; ++i) {           // Jacobi recurrence, a=b=1
                float fi   = (float)i;
                float thk  = ((2.0f * fi + 2.0f) * (2.0f * fi + 1.0f)) /
                             (2.0f * fi * (fi + 2.0f));
                float thk2 = (fi + 1.0f) / (fi + 2.0f);
                float p = thk * t * pm1 - thk2 * pm2;
                act[(i * BM + r) * ACT_STRIDE + ac] = f2bf(p);
                pm2 = pm1; pm1 = p;
            }
        }
        __syncthreads();

        // ---- 9 branches x (2x4) WMMA, pointer-bumped B addressing ----
        const uint4* bp = wb + (size_t)ic * 4096 + bWave;
#pragma unroll 3
        for (int d = 0; d < NBR; ++d) {
            FragAB a[2];
#pragma unroll
            for (int mi = 0; mi < 2; ++mi) {
                const unsigned short* base =
                    &act[(d * BM + arow + mi * 16) * ACT_STRIDE];
                a[mi].u[0] = *reinterpret_cast<const uint4*>(base + kcol);
                a[mi].u[1] = *reinterpret_cast<const uint4*>(base + kcol + 8);
            }
            FragAB b[4];
#pragma unroll
            for (int nj = 0; nj < 4; ++nj) {
                b[nj].u[0] = bp[nj * 64];
                b[nj].u[1] = bp[nj * 64 + 32];
            }
#pragma unroll
            for (int mi = 0; mi < 2; ++mi)
#pragma unroll
                for (int nj = 0; nj < 4; ++nj)
                    acc[mi][nj] = __builtin_amdgcn_wmma_f32_16x16x32_bf16(
                        false, a[mi].v, false, b[nj].v,
                        (short)0, acc[mi][nj], false, false);
            bp += 32 * 4096;    // next branch: kt += 32
        }
        __syncthreads();
    }

    // ---- store z (C layout: VGPR v -> M = v + 8*(lane>=16), N = lane%16) ----
    const int mOff = (lane >> 4) * 8;
    const int nOff = lane & 15;
#pragma unroll
    for (int mi = 0; mi < 2; ++mi)
#pragma unroll
        for (int nj = 0; nj < 4; ++nj) {
            int m0 = rowBase + wm * 32 + mi * 16 + mOff;
            int n0 = ntBase * 16 + wn * 64 + nj * 16 + nOff;
#pragma unroll
            for (int v = 0; v < 8; ++v)
                z[(size_t)(m0 + v) * OUT_F + n0] = acc[mi][nj][v];
        }
}

// ---------------------------------------------------------------------------
// Kernel 4: in-place (z + jbias) -> LayerNorm -> SiLU per row.
// ---------------------------------------------------------------------------
__global__ __launch_bounds__(256) void kan_ln_silu(
    float* __restrict__ z, const float* __restrict__ jb,
    const float* __restrict__ lnw, const float* __restrict__ lnb)
{
    __shared__ float ws1[8], ws2[8];
    const int lane = threadIdx.x & 31;
    const int wave = threadIdx.x >> 5;
    float* p = z + (size_t)blockIdx.x * OUT_F;

    float vals[4], s = 0.0f, s2 = 0.0f;
#pragma unroll
    for (int j = 0; j < 4; ++j) {
        int c = threadIdx.x + j * 256;
        float v = p[c] + jb[c];
        vals[j] = v; s += v; s2 += v * v;
    }
#pragma unroll
    for (int off = 16; off > 0; off >>= 1) {
        s  += __shfl_xor(s,  off, 32);
        s2 += __shfl_xor(s2, off, 32);
    }
    if (lane == 0) { ws1[wave] = s; ws2[wave] = s2; }
    __syncthreads();
    float S = 0.0f, S2 = 0.0f;
#pragma unroll
    for (int w = 0; w < 8; ++w) { S += ws1[w]; S2 += ws2[w]; }

    const float mu  = S * (1.0f / OUT_F);
    const float var = S2 * (1.0f / OUT_F) - mu * mu;
    const float inv = rsqrtf(var + 1e-5f);
#pragma unroll
    for (int j = 0; j < 4; ++j) {
        int c = threadIdx.x + j * 256;
        float zh = (vals[j] - mu) * inv * lnw[c] + lnb[c];
        p[c] = fast_silu(zh);
    }
}

// ---------------------------------------------------------------------------
extern "C" void kernel_launch(void* const* d_in, const int* in_sizes, int n_in,
                              void* d_out, int out_size, void* d_ws, size_t ws_size,
                              hipStream_t stream) {
    const float* x   = (const float*)d_in[0];   // [8192][1024]
    const float* bw  = (const float*)d_in[1];   // [1024][1024]
    const float* jc  = (const float*)d_in[2];   // [1024][1024][9]
    const float* lnw = (const float*)d_in[3];   // [1024]
    const float* lnb = (const float*)d_in[4];   // [1024]
    float* out = (float*)d_out;                 // [8192][1024]

    uint4* wb  = (uint4*)d_ws;                          // 288*4096 uint4 = 18.9 MB
    float* jb  = (float*)((char*)d_ws + (size_t)KTILES * 4096 * sizeof(uint4));

    // 1) pack weights into WMMA-fragment-native bf16
    const unsigned nPack = KTILES * 4096;               // 1,179,648 threads
    kan_pack_b<<<nPack / 256, 256, 0, stream>>>(bw, jc, wb);

    // 2) P0-branch bias vector
    kan_jbias<<<OUT_F / 256, 256, 0, stream>>>(jc, jb);

    // 3) fused activation + 9-branch bf16 WMMA GEMM (90 KB dynamic LDS)
    dim3 grid(BATCH / BM, OUT_F / 128);
    kan_gemm<<<grid, 256, ACT_BYTES, stream>>>(x, wb, out);

    // 4) bias + LayerNorm + SiLU, in place on d_out
    kan_ln_silu<<<BATCH, 256, 0, stream>>>(out, jb, lnw, lnb);
}
